// SparseMoE_22316650070634
// MI455X (gfx1250) — matrix operand order repaired and checked by reading, not compiled, not run on hardware
//
#include <hip/hip_runtime.h>
#include <hip/hip_bf16.h>

#define D_EMBED 1024
#define N_EXP 8
#define H_DIM 4096
#define N_TOK 4096            // B*S
#define N_ASSIGN (N_TOK * 2)
#define MAXROWS 9216          // 8192 assignments + per-expert 128-alignment pad
#define MAXTILES 72           // MAXROWS / 128
#define TILE 128
#define KSTEP 32
#define LDS_STRIDE 40         // ushort stride per LDS row (32 data + 8 pad; 80B, 16B-aligned)

typedef __attribute__((ext_vector_type(16))) __bf16 v16bf;
typedef __attribute__((ext_vector_type(8)))  float  v8f;

union FragU { uint4 q[2]; v16bf v; };

// round-to-nearest-even fp32 -> bf16 (epilogue only)
__device__ __forceinline__ unsigned short f2bf(float f) {
  unsigned int u = __float_as_uint(f);
  u += 0x7FFFu + ((u >> 16) & 1u);
  return (unsigned short)(u >> 16);
}

// pack two fp32 into bf16x2 with a single v_perm_b32 (truncate)
__device__ __forceinline__ unsigned int pack_bf16(float f0, float f1) {
  return __builtin_amdgcn_perm(__float_as_uint(f1), __float_as_uint(f0), 0x07060302u);
}

// generic->LDS offset (low 32 bits of the flat address are the LDS offset)
__device__ __forceinline__ unsigned int lds_off(const void* p) {
  return (unsigned int)(unsigned long long)p;
}

// CDNA5 async copy: 16B global -> LDS, tracked by ASYNCcnt
__device__ __forceinline__ void async_copy_b128(unsigned int dst_lds, const void* gsrc) {
  asm volatile("global_load_async_to_lds_b128 %0, %1, off"
               :: "v"(dst_lds), "v"((unsigned long long)gsrc)
               : "memory");
}

__device__ __forceinline__ void wait_async0() {
#if __has_builtin(__builtin_amdgcn_s_wait_asynccnt)
  __builtin_amdgcn_s_wait_asynccnt(0);
#else
  asm volatile("s_wait_asynccnt 0x0" ::: "memory");
#endif
}

// ---------------- prep: x -> bf16, zero output ----------------
__global__ void moe_prep(const float* __restrict__ x,
                         unsigned short* __restrict__ Xbf,
                         float* __restrict__ out, int n) {
  int i = blockIdx.x * 256 + threadIdx.x;
  if (i < n) { Xbf[i] = f2bf(x[i]); out[i] = 0.f; }
}

// ---------------- router: relu(x@Wr + br), top-2 masked softmax ----------------
__global__ __launch_bounds__(256) void moe_router(const float* __restrict__ x,
                                                  const float* __restrict__ Wr,
                                                  const float* __restrict__ br,
                                                  int* __restrict__ topE,
                                                  float* __restrict__ topW) {
  __shared__ float red[256];
  __shared__ float sc[N_EXP];
  int t = blockIdx.x, tid = threadIdx.x;
  const float* xr = x + (size_t)t * D_EMBED;
  float acc[N_EXP] = {0.f, 0.f, 0.f, 0.f, 0.f, 0.f, 0.f, 0.f};
  for (int d = tid; d < D_EMBED; d += 256) {
    float xv = xr[d];
    const float* wr = Wr + (size_t)d * N_EXP;
#pragma unroll
    for (int e = 0; e < N_EXP; ++e) acc[e] += xv * wr[e];
  }
  for (int e = 0; e < N_EXP; ++e) {
    red[tid] = acc[e];
    __syncthreads();
    for (int s = 128; s > 0; s >>= 1) {
      if (tid < s) red[tid] += red[tid + s];
      __syncthreads();
    }
    if (tid == 0) sc[e] = red[0];
    __syncthreads();
  }
  if (tid == 0) {
    float s[N_EXP];
#pragma unroll
    for (int e = 0; e < N_EXP; ++e) {
      float v = sc[e] + br[e];
      s[e] = v > 0.f ? v : 0.f;
    }
    int i1 = 0;
    for (int e = 1; e < N_EXP; ++e) if (s[e] > s[i1]) i1 = e;
    int i2 = (i1 == 0) ? 1 : 0;
    for (int e = 0; e < N_EXP; ++e) if (e != i1 && s[e] > s[i2]) i2 = e;
    float e2 = __expf(s[i2] - s[i1]);     // s[i1] >= s[i2]
    float inv = 1.f / (1.f + e2);
    topE[t * 2 + 0] = i1; topW[t * 2 + 0] = inv;
    topE[t * 2 + 1] = i2; topW[t * 2 + 1] = e2 * inv;
  }
}

// ---------------- compact: per-expert row lists, 128-aligned bases ----------------
__global__ void moe_compact(const int* __restrict__ topE, const float* __restrict__ topW,
                            int* __restrict__ tileExpert,
                            int* __restrict__ rowToken, float* __restrict__ rowW) {
  __shared__ int cnt[N_EXP], base[N_EXP], fill[N_EXP];
  int tid = threadIdx.x;
  if (tid < N_EXP) { cnt[tid] = 0; fill[tid] = 0; }
  __syncthreads();
  for (int i = tid; i < N_ASSIGN; i += 256) atomicAdd(&cnt[topE[i]], 1);
  __syncthreads();
  if (tid == 0) {
    int b = 0;
    for (int e = 0; e < N_EXP; ++e) {
      base[e] = b;
      int tiles = (cnt[e] + TILE - 1) / TILE;
      for (int m = 0; m < tiles; ++m) tileExpert[b / TILE + m] = e;
      b += tiles * TILE;
    }
    for (int m = b / TILE; m < MAXTILES; ++m) tileExpert[m] = -1;
  }
  __syncthreads();
  for (int i = tid; i < MAXROWS; i += 256) { rowToken[i] = 0; rowW[i] = 0.f; }
  __syncthreads();
  for (int i = tid; i < N_ASSIGN; i += 256) {
    int e = topE[i];
    int pos = atomicAdd(&fill[e], 1);
    int gr = base[e] + pos;
    rowToken[gr] = i >> 1;
    rowW[gr] = topW[i];
  }
}

// ---------------- GEMM1: h = relu(Xg @ W1[e] + b1[e]) -> bf16 ----------------
__global__ __launch_bounds__(256, 1) void moe_gemm1(
    const unsigned short* __restrict__ Xbf, const float* __restrict__ W1,
    const float* __restrict__ b1, const int* __restrict__ rowToken,
    const int* __restrict__ tileExpert, unsigned short* __restrict__ Hbuf) {
  int mt = blockIdx.y, nt = blockIdx.x;
  int e = tileExpert[mt];
  if (e < 0) return;
  const float* W1e = W1 + (size_t)e * D_EMBED * H_DIM;
  const float* b1e = b1 + (size_t)e * H_DIM;

  __shared__ unsigned short As[2][TILE * LDS_STRIDE];
  __shared__ unsigned short Bs[2][TILE * LDS_STRIDE];

  int tid = threadIdx.x;
  int lane = tid & 31, wid = tid >> 5;
  int laneHi = lane >> 4, lrow = lane & 15;
  int m0w = (wid & 1) * 64, n0w = (wid >> 1) * 32;
  int n0 = nt * TILE;

  int tokens[2];
#pragma unroll
  for (int it = 0; it < 2; ++it)
    tokens[it] = rowToken[mt * TILE + ((tid + it * 256) >> 2)];

  v8f acc[4][2];
#pragma unroll
  for (int i = 0; i < 4; ++i)
#pragma unroll
    for (int j = 0; j < 2; ++j)
      acc[i][j] = (v8f){0.f, 0.f, 0.f, 0.f, 0.f, 0.f, 0.f, 0.f};

  unsigned int bReg[8];

  // async gather of A rows (already bf16): global -> LDS, no registers
  auto stageA = [&](int b, int kk) {
    int k0 = kk * KSTEP;
#pragma unroll
    for (int it = 0; it < 2; ++it) {
      int j = tid + it * 256;          // 0..511
      int r = j >> 2, c = j & 3;       // 128 rows x 4 chunks of 8 bf16
      async_copy_b128(lds_off(&As[b][r * LDS_STRIDE + c * 8]),
                      Xbf + (size_t)tokens[it] * D_EMBED + k0 + c * 8);
    }
  };
  auto stageBload = [&](int kk) {
    int k0 = kk * KSTEP;
#pragma unroll
    for (int it = 0; it < 8; ++it) {
      int j = tid + it * 256, n = j & 127, kp = j >> 7;   // kp 0..15
      float f0 = W1e[(size_t)(k0 + 2 * kp + 0) * H_DIM + n0 + n];
      float f1 = W1e[(size_t)(k0 + 2 * kp + 1) * H_DIM + n0 + n];
      bReg[it] = pack_bf16(f0, f1);
    }
    if (k0 + KSTEP < D_EMBED)
      __builtin_prefetch(&W1e[(size_t)(k0 + KSTEP) * H_DIM + n0 + (tid & 127)], 0, 1);
  };
  auto stageBstore = [&](int b) {
#pragma unroll
    for (int it = 0; it < 8; ++it) {
      int j = tid + it * 256, n = j & 127, kp = j >> 7;
      *(unsigned int*)&Bs[b][n * LDS_STRIDE + 2 * kp] = bReg[it];
    }
  };
  auto compute = [&](int b) {
    FragU afr[4], bfr[2];
    int akb = laneHi * 16;               // A: 16 contiguous K per lane-half
#pragma unroll
    for (int i = 0; i < 4; ++i) {
      const unsigned short* p = &As[b][(m0w + i * 16 + lrow) * LDS_STRIDE + akb];
      afr[i].q[0] = *(const uint4*)p;
      afr[i].q[1] = *(const uint4*)(p + 8);
    }
    int bkb = laneHi * 16;               // B: 16 contiguous K per lane-half
#pragma unroll
    for (int i = 0; i < 2; ++i) {
      const unsigned short* p = &Bs[b][(n0w + i * 16 + lrow) * LDS_STRIDE + bkb];
      bfr[i].q[0] = *(const uint4*)p;
      bfr[i].q[1] = *(const uint4*)(p + 8);
    }
#pragma unroll
    for (int i = 0; i < 4; ++i)
#pragma unroll
      for (int jn = 0; jn < 2; ++jn)
        acc[i][jn] = __builtin_amdgcn_wmma_f32_16x16x32_bf16(
            false, afr[i].v, false, bfr[jn].v, (short)0, acc[i][jn], false, false);
  };

  stageA(0, 0);
  stageBload(0);
  stageBstore(0);
  wait_async0();
  __syncthreads();
  const int NK = D_EMBED / KSTEP;        // 32
  for (int kk = 0; kk < NK; ++kk) {
    int b = kk & 1;
    if (kk + 1 < NK) { stageA(b ^ 1, kk + 1); stageBload(kk + 1); }
    compute(b);
    if (kk + 1 < NK) stageBstore(b ^ 1);
    wait_async0();
    __syncthreads();
  }

#pragma unroll
  for (int jn = 0; jn < 2; ++jn) {
    int col = n0 + n0w + jn * 16 + lrow;
    float bias = b1e[col];
#pragma unroll
    for (int i = 0; i < 4; ++i)
#pragma unroll
      for (int j = 0; j < 8; ++j) {
        int r = m0w + i * 16 + laneHi * 8 + j;
        float v = acc[i][jn][j] + bias;
        v = v > 0.f ? v : 0.f;
        Hbuf[(size_t)(mt * TILE + r) * H_DIM + col] = f2bf(v);
      }
  }
}

// ---------------- GEMM2: out[t] += w * (h @ W2[e] + b2[e]) ----------------
__global__ __launch_bounds__(256, 1) void moe_gemm2(
    const unsigned short* __restrict__ Hbuf, const float* __restrict__ W2,
    const float* __restrict__ b2, const int* __restrict__ rowToken,
    const float* __restrict__ rowW, const int* __restrict__ tileExpert,
    float* __restrict__ out) {
  int mt = blockIdx.y, nt = blockIdx.x;
  int e = tileExpert[mt];
  if (e < 0) return;
  const float* W2e = W2 + (size_t)e * H_DIM * D_EMBED;
  const float* b2e = b2 + (size_t)e * D_EMBED;

  __shared__ unsigned short As[2][TILE * LDS_STRIDE];
  __shared__ unsigned short Bs[2][TILE * LDS_STRIDE];

  int tid = threadIdx.x;
  int lane = tid & 31, wid = tid >> 5;
  int laneHi = lane >> 4, lrow = lane & 15;
  int m0w = (wid & 1) * 64, n0w = (wid >> 1) * 32;
  int n0 = nt * TILE;

  v8f acc[4][2];
#pragma unroll
  for (int i = 0; i < 4; ++i)
#pragma unroll
    for (int j = 0; j < 2; ++j)
      acc[i][j] = (v8f){0.f, 0.f, 0.f, 0.f, 0.f, 0.f, 0.f, 0.f};

  unsigned int bReg[8];

  auto stageA = [&](int b, int kk) {
    int k0 = kk * KSTEP;
#pragma unroll
    for (int it = 0; it < 2; ++it) {
      int j = tid + it * 256;
      int r = j >> 2, c = j & 3;
      async_copy_b128(lds_off(&As[b][r * LDS_STRIDE + c * 8]),
                      Hbuf + (size_t)(mt * TILE + r) * H_DIM + k0 + c * 8);
    }
  };
  auto stageBload = [&](int kk) {
    int k0 = kk * KSTEP;
#pragma unroll
    for (int it = 0; it < 8; ++it) {
      int j = tid + it * 256, n = j & 127, kp = j >> 7;
      float f0 = W2e[(size_t)(k0 + 2 * kp + 0) * D_EMBED + n0 + n];
      float f1 = W2e[(size_t)(k0 + 2 * kp + 1) * D_EMBED + n0 + n];
      bReg[it] = pack_bf16(f0, f1);
    }
    if (k0 + KSTEP < H_DIM)
      __builtin_prefetch(&W2e[(size_t)(k0 + KSTEP) * D_EMBED + n0 + (tid & 127)], 0, 1);
  };
  auto stageBstore = [&](int b) {
#pragma unroll
    for (int it = 0; it < 8; ++it) {
      int j = tid + it * 256, n = j & 127, kp = j >> 7;
      *(unsigned int*)&Bs[b][n * LDS_STRIDE + 2 * kp] = bReg[it];
    }
  };
  auto compute = [&](int b) {
    FragU afr[4], bfr[2];
    int akb = laneHi * 16;
#pragma unroll
    for (int i = 0; i < 4; ++i) {
      const unsigned short* p = &As[b][(m0w + i * 16 + lrow) * LDS_STRIDE + akb];
      afr[i].q[0] = *(const uint4*)p;
      afr[i].q[1] = *(const uint4*)(p + 8);
    }
    int bkb = laneHi * 16;
#pragma unroll
    for (int i = 0; i < 2; ++i) {
      const unsigned short* p = &Bs[b][(n0w + i * 16 + lrow) * LDS_STRIDE + bkb];
      bfr[i].q[0] = *(const uint4*)p;
      bfr[i].q[1] = *(const uint4*)(p + 8);
    }
#pragma unroll
    for (int i = 0; i < 4; ++i)
#pragma unroll
      for (int jn = 0; jn < 2; ++jn)
        acc[i][jn] = __builtin_amdgcn_wmma_f32_16x16x32_bf16(
            false, afr[i].v, false, bfr[jn].v, (short)0, acc[i][jn], false, false);
  };

  stageA(0, 0);
  stageBload(0);
  stageBstore(0);
  wait_async0();
  __syncthreads();
  const int NK = H_DIM / KSTEP;            // 128
  for (int kk = 0; kk < NK; ++kk) {
    int b = kk & 1;
    if (kk + 1 < NK) { stageA(b ^ 1, kk + 1); stageBload(kk + 1); }
    compute(b);
    if (kk + 1 < NK) stageBstore(b ^ 1);
    wait_async0();
    __syncthreads();
  }

#pragma unroll
  for (int jn = 0; jn < 2; ++jn) {
    int col = n0 + n0w + jn * 16 + lrow;
    float b2v = b2e[col];
#pragma unroll
    for (int i = 0; i < 4; ++i)
#pragma unroll
      for (int j = 0; j < 8; ++j) {
        int r = m0w + i * 16 + laneHi * 8 + j;
        int gr = mt * TILE + r;
        float w = rowW[gr];
        if (w != 0.f) {
          int t = rowToken[gr];
          atomicAdd(&out[(size_t)t * D_EMBED + col], w * (acc[i][jn][j] + b2v));
        }
      }
  }
}

// ---------------- launch ----------------
extern "C" void kernel_launch(void* const* d_in, const int* in_sizes, int n_in,
                              void* d_out, int out_size, void* d_ws, size_t ws_size,
                              hipStream_t stream) {
  const float* x  = (const float*)d_in[0];
  const float* W1 = (const float*)d_in[1];
  const float* b1 = (const float*)d_in[2];
  const float* W2 = (const float*)d_in[3];
  const float* b2 = (const float*)d_in[4];
  const float* Wr = (const float*)d_in[5];
  const float* br = (const float*)d_in[6];
  float* out = (float*)d_out;

  char* w = (char*)d_ws;
  size_t off = 0;
  unsigned short* Hbuf = (unsigned short*)(w + off); off += (size_t)MAXROWS * H_DIM * 2;
  unsigned short* Xbf  = (unsigned short*)(w + off); off += (size_t)N_TOK * D_EMBED * 2;
  int*   rowToken = (int*)(w + off);   off += (size_t)MAXROWS * 4;
  float* rowW     = (float*)(w + off); off += (size_t)MAXROWS * 4;
  int*   topE     = (int*)(w + off);   off += (size_t)N_ASSIGN * 4;
  float* topW     = (float*)(w + off); off += (size_t)N_ASSIGN * 4;
  int*   tileExpert = (int*)(w + off); off += (size_t)MAXTILES * 4 + 256;

  int n = N_TOK * D_EMBED;
  moe_prep<<<(n + 255) / 256, 256, 0, stream>>>(x, Xbf, out, n);
  moe_router<<<N_TOK, 256, 0, stream>>>(x, Wr, br, topE, topW);
  moe_compact<<<1, 256, 0, stream>>>(topE, topW, tileExpert, rowToken, rowW);
  moe_gemm1<<<dim3(H_DIM / TILE, MAXTILES), 256, 0, stream>>>(Xbf, W1, b1, rowToken, tileExpert, Hbuf);
  moe_gemm2<<<dim3(D_EMBED / TILE, MAXTILES), 256, 0, stream>>>(Hbuf, W2, b2, rowToken, rowW, tileExpert, out);
}
